// InferDecoder_16956530884971
// MI455X (gfx1250) — matrix operand order repaired and checked by reading, not compiled, not run on hardware
//
#include <hip/hip_runtime.h>
#include <hip/hip_bf16.h>
#include <math.h>

typedef __attribute__((ext_vector_type(16))) __bf16 v16bf;
typedef __attribute__((ext_vector_type(8)))  __bf16 v8bf;
typedef __attribute__((ext_vector_type(8)))  float  v8f;
typedef __attribute__((ext_vector_type(4)))  int    v4i;

union FragA { v16bf v; v8bf h[2]; };

#define B_  64
#define H_  1024
#define E_  1024
#define V_  32000
#define T_  32

#define CHUNK_K     128                    // K depth staged per LDS chunk
#define CHUNK_ELEMS (CHUNK_K * 64)         // 8192 bf16 = 16 KB

// ---------------------------------------------------------------------------
// CDNA5 async global->LDS copy (ASYNCcnt path), guarded with fallback.
// Builtin signature (probe-confirmed by diagnostics): 
//   (v4i addrspace(1)* gsrc, v4i addrspace(3)* ldst, imm offset, imm cpol)
// ---------------------------------------------------------------------------
#if defined(__HIP_DEVICE_COMPILE__) && __has_builtin(__builtin_amdgcn_global_load_async_to_lds_b128)
#define HAVE_ASYNC_LDS 1
#else
#define HAVE_ASYNC_LDS 0
#endif

__device__ __forceinline__ void async_copy16(const __bf16* gsrc, __bf16* ldst)
{
#if HAVE_ASYNC_LDS
    __builtin_amdgcn_global_load_async_to_lds_b128(
        (__attribute__((address_space(1))) v4i*)gsrc,
        (__attribute__((address_space(3))) v4i*)ldst,
        0, 0);
#else
    *(v8bf*)ldst = *(const v8bf*)gsrc;     // fallback: global load + ds_store
#endif
}

template <int N>
__device__ __forceinline__ void wait_asynccnt()
{
#if defined(__HIP_DEVICE_COMPILE__)
#if __has_builtin(__builtin_amdgcn_s_wait_asynccnt)
    __builtin_amdgcn_s_wait_asynccnt(N);
#else
    asm volatile("s_wait_asynccnt %0" :: "i"(N) : "memory");
#endif
#endif
}

// ---------------------------------------------------------------------------
// A-fragment swizzle for activations: element (row, k) of a [64, K] bf16
// matrix is stored so that each WMMA A-fragment (one lane's 16 bf16 of a
// 16x32 tile) is contiguous, and each 32-K block of all four M-tiles is a
// contiguous 4 KB run (async-copy friendly).
//   block = bk*4 + mt   (bk = k/32, mt = row/16), 512 bf16 per block
//   lane  = (row%16) + 16*bit3(k%32)
//   idx   = (k%8) | (bit4(k%32) << 3)
// ---------------------------------------------------------------------------
__device__ __forceinline__ size_t xh_swz_off(int row, int k)
{
    int bk   = k >> 5;
    int kin  = k & 31;
    int mt   = row >> 4;
    int l15  = row & 15;
    int lane = l15 + (((kin >> 3) & 1) << 4);
    int idx  = (kin & 7) | (((kin >> 4) & 1) << 3);
    return ((size_t)(bk * 4 + mt) * 32 + lane) * 16 + idx;
}

// ---------------------------------------------------------------------------
// Convert f32 weights [Wa;Wb] (K x N, k-major) into the WMMA B-fragment
// swizzled bf16 layout: block(bn,bk) = 512 bf16, lane-major, 16 contiguous
// bf16 per lane (lanes 0-15: K=bk*32+{0-7,16-23} of col bn*16+lane; lanes 16-31:
// K=+{8-15,24-31}), matching V_WMMA_*_16X16X32 B-matrix striping.
// ---------------------------------------------------------------------------
__global__ void convert_weights_swz(const float* __restrict__ Wa,
                                    const float* __restrict__ Wb,
                                    int ksplit, int K, int N,
                                    __bf16* __restrict__ dst)
{
    size_t i = (size_t)blockIdx.x * blockDim.x + threadIdx.x;
    size_t total = (size_t)K * N;
    if (i >= total) return;
    int idx    = (int)(i & 15);
    int lane   = (int)((i >> 4) & 31);
    size_t blk = i >> 9;
    int kblocks = K >> 5;
    int bk = (int)(blk % (size_t)kblocks);
    int bn = (int)(blk / (size_t)kblocks);
    int n = bn * 16 + (lane & 15);
    int k = bk * 32 + ((lane >> 4) ? 8 : 0) + (idx & 7) + ((idx >> 3) << 4);
    float v = (k < ksplit) ? Wa[(size_t)k * N + n]
                           : Wb[(size_t)(k - ksplit) * N + n];
    dst[i] = (__bf16)v;
}

// ---------------------------------------------------------------------------
// WMMA GEMM: Out[64, N] = A[64, K] @ W[K, N]
//   A: bf16, fragment-swizzled (xh_swz_off layout), staged to LDS per-WG via
//      async copies (double-buffered 16 KB chunks, shared by 8 waves).
//   W: bf16, B-fragment swizzled, streamed global->VGPR with prefetch.
// One wave per 16-wide N strip; each wave does all four 16x16 M-tiles.
// Grid must satisfy N == gridDim.x * 128.
// ---------------------------------------------------------------------------
__global__ void __launch_bounds__(256)
gemm_wmma_bf16(const __bf16* __restrict__ Aswz,
               const __bf16* __restrict__ Wswz,
               float* __restrict__ Out, int N, int K)
{
    __shared__ __align__(16) __bf16 abuf[2][CHUNK_ELEMS];

    const int tid  = threadIdx.x;
    const int lane = tid & 31;
    const int wave = tid >> 5;
    const int bn   = blockIdx.x * 8 + wave;       // n-tile owned by this wave
    const int nchunks = K / CHUNK_K;

    // ---- prologue: issue async copies for chunks 0 and 1 ----
    {
        const __bf16* g0 = Aswz + (size_t)tid * 8;
        __bf16*       l0 = &abuf[0][(size_t)tid * 8];
        #pragma unroll
        for (int i = 0; i < 4; ++i)
            async_copy16(g0 + i * 2048, l0 + i * 2048);
        const __bf16* g1 = Aswz + CHUNK_ELEMS + (size_t)tid * 8;
        __bf16*       l1 = &abuf[1][(size_t)tid * 8];
        #pragma unroll
        for (int i = 0; i < 4; ++i)
            async_copy16(g1 + i * 2048, l1 + i * 2048);
    }

    v8f acc0 = {}, acc1 = {}, acc2 = {}, acc3 = {};
    const __bf16* wp = Wswz + ((size_t)bn * (size_t)(K >> 5)) * 512 + (size_t)lane * 16;

    for (int c = 0; c < nchunks; ++c) {
        // own chunk-c copies complete (4 in-order b128s per newer chunk)
        if (c + 1 < nchunks) wait_asynccnt<4>(); else wait_asynccnt<0>();
        __syncthreads();                          // all waves' copies visible

        const __bf16* lbase = abuf[c & 1];
        #pragma unroll
        for (int kb = 0; kb < 4; ++kb) {          // 4 k-blocks of 32 per chunk
            v16bf bfrag = *(const v16bf*)wp;      // 32B/lane weight fragment
            wp += 512;
            __builtin_prefetch(wp, 0, 1);         // global_prefetch weight stream

            const __bf16* ab = lbase + ((size_t)(kb * 4) * 32 + lane) * 16;
            FragA a0, a1, a2, a3;                 // ds_load_b128 pairs from LDS
            a0.h[0] = *(const v8bf*)(ab);        a0.h[1] = *(const v8bf*)(ab + 8);
            ab += 512;
            a1.h[0] = *(const v8bf*)(ab);        a1.h[1] = *(const v8bf*)(ab + 8);
            ab += 512;
            a2.h[0] = *(const v8bf*)(ab);        a2.h[1] = *(const v8bf*)(ab + 8);
            ab += 512;
            a3.h[0] = *(const v8bf*)(ab);        a3.h[1] = *(const v8bf*)(ab + 8);

            acc0 = __builtin_amdgcn_wmma_f32_16x16x32_bf16(false, a0.v, false, bfrag,
                                                           (short)0, acc0, false, false);
            acc1 = __builtin_amdgcn_wmma_f32_16x16x32_bf16(false, a1.v, false, bfrag,
                                                           (short)0, acc1, false, false);
            acc2 = __builtin_amdgcn_wmma_f32_16x16x32_bf16(false, a2.v, false, bfrag,
                                                           (short)0, acc2, false, false);
            acc3 = __builtin_amdgcn_wmma_f32_16x16x32_bf16(false, a3.v, false, bfrag,
                                                           (short)0, acc3, false, false);
        }
        __syncthreads();                          // all waves done reading buffer

        if (c + 2 < nchunks) {                    // refill freed buffer
            const __bf16* g = Aswz + (size_t)(c + 2) * CHUNK_ELEMS + (size_t)tid * 8;
            __bf16*       l = &abuf[c & 1][(size_t)tid * 8];
            #pragma unroll
            for (int i = 0; i < 4; ++i)
                async_copy16(g + i * 2048, l + i * 2048);
        }
    }

    // C/D layout: VGPR r -> (lanes 0-15: M=r, lanes 16-31: M=r+8), N=lane%16
    const int col   = bn * 16 + (lane & 15);
    const int rbase = (lane >> 4) ? 8 : 0;
    #pragma unroll
    for (int r = 0; r < 8; ++r) {
        Out[(size_t)(rbase + r)      * N + col] = acc0[r];
        Out[(size_t)(16 + rbase + r) * N + col] = acc1[r];
        Out[(size_t)(32 + rbase + r) * N + col] = acc2[r];
        Out[(size_t)(48 + rbase + r) * N + col] = acc3[r];
    }
}

// ---------------------------------------------------------------------------
// State init: c (f32) into workspace, h -> bf16 into xh h-slot (swizzled)
// ---------------------------------------------------------------------------
__global__ void init_layer_state(const float* __restrict__ h0,
                                 const float* __restrict__ c0,
                                 __bf16* __restrict__ xh,
                                 float* __restrict__ c_state)
{
    int i = blockIdx.x * blockDim.x + threadIdx.x;   // 64*1024
    int b = i >> 10, j = i & 1023;
    xh[xh_swz_off(b, 1024 + j)] = (__bf16)h0[i];
    c_state[i] = c0[i];
}

__global__ void init_tok(int* __restrict__ tok)
{
    if (threadIdx.x < B_) tok[threadIdx.x] = 1;      // START token
}

__global__ void embed_gather(const float* __restrict__ emb,
                             const int* __restrict__ tok,
                             __bf16* __restrict__ xh0)
{
    int i = blockIdx.x * blockDim.x + threadIdx.x;   // 64*1024
    int b = i >> 10, j = i & 1023;
    xh0[xh_swz_off(b, j)] = (__bf16)emb[(size_t)tok[b] * E_ + j];
}

// ---------------------------------------------------------------------------
// Fused LSTM gates: z[64,4096] (+bias) -> c update; h -> bf16 into this
// layer's next-step h-slot and the next consumer's x-slot (both swizzled).
// ---------------------------------------------------------------------------
__global__ void lstm_gate(const float* __restrict__ z,
                          const float* __restrict__ bias,
                          float* __restrict__ c_state,
                          __bf16* __restrict__ xh_self,
                          __bf16* __restrict__ x_next)
{
    int i = blockIdx.x * blockDim.x + threadIdx.x;   // 64*1024
    int b = i >> 10, j = i & 1023;
    const float* zr = z + (size_t)b * 4096;
    float ig = 1.f / (1.f + expf(-(zr[j]        + bias[j])));
    float fg = 1.f / (1.f + expf(-(zr[1024 + j] + bias[1024 + j])));
    float gg = tanhf(zr[2048 + j] + bias[2048 + j]);
    float og = 1.f / (1.f + expf(-(zr[3072 + j] + bias[3072 + j])));
    float c  = fg * c_state[i] + ig * gg;
    c_state[i] = c;
    float hh = og * tanhf(c);
    __bf16 hb = (__bf16)hh;
    xh_self[xh_swz_off(b, 1024 + j)] = hb;           // h input at t+1
    x_next[xh_swz_off(b, j)]         = hb;           // x input of next GEMM
}

// ---------------------------------------------------------------------------
// Softmax + argmax per batch row; writes probs[B,T,V] f32 and int64 tokens.
// ---------------------------------------------------------------------------
__global__ void softmax_argmax(const float* __restrict__ logits,
                               const float* __restrict__ bfc,
                               float* __restrict__ probs,
                               long long* __restrict__ toks_out,
                               int* __restrict__ tok_next, int t)
{
    __shared__ float sv[256];
    __shared__ int   si[256];
    const int b   = blockIdx.x;
    const int tid = threadIdx.x;
    const float* lrow = logits + (size_t)b * V_;

    float bv = -3.402823466e+38f; int bi = V_;
    for (int v = tid; v < V_; v += 256) {
        float x = lrow[v] + bfc[v];
        if (x > bv) { bv = x; bi = v; }
    }
    sv[tid] = bv; si[tid] = bi;
    __syncthreads();
    for (int s = 128; s > 0; s >>= 1) {
        if (tid < s) {
            float ov = sv[tid + s]; int oi = si[tid + s];
            if (ov > sv[tid] || (ov == sv[tid] && oi < si[tid])) {
                sv[tid] = ov; si[tid] = oi;
            }
        }
        __syncthreads();
    }
    const float rowmax = sv[0];
    const int   amax   = si[0];
    __syncthreads();

    float lsum = 0.f;
    for (int v = tid; v < V_; v += 256) lsum += expf(lrow[v] + bfc[v] - rowmax);
    sv[tid] = lsum;
    __syncthreads();
    for (int s = 128; s > 0; s >>= 1) {
        if (tid < s) sv[tid] += sv[tid + s];
        __syncthreads();
    }
    const float inv = 1.f / sv[0];

    float* prow = probs + ((size_t)b * T_ + t) * (size_t)V_;
    for (int v = tid; v < V_; v += 256)
        prow[v] = expf(lrow[v] + bfc[v] - rowmax) * inv;

    if (tid == 0) {
        tok_next[b] = amax;
        toks_out[(size_t)b * T_ + t] = (long long)amax;
    }
}

// ---------------------------------------------------------------------------
extern "C" void kernel_launch(void* const* d_in, const int* in_sizes, int n_in,
                              void* d_out, int out_size, void* d_ws, size_t ws_size,
                              hipStream_t stream)
{
    (void)in_sizes; (void)n_in; (void)out_size; (void)ws_size;

    const float* h_in[4] = { (const float*)d_in[0], (const float*)d_in[2],
                             (const float*)d_in[4], (const float*)d_in[6] };
    const float* c_in[4] = { (const float*)d_in[1], (const float*)d_in[3],
                             (const float*)d_in[5], (const float*)d_in[7] };
    const float* emb = (const float*)d_in[8];
    const float *W[4], *U[4], *bias[4];
    for (int l = 0; l < 4; ++l) {
        W[l]    = (const float*)d_in[9  + 3 * l];
        U[l]    = (const float*)d_in[10 + 3 * l];
        bias[l] = (const float*)d_in[11 + 3 * l];
    }
    const float* Wfc = (const float*)d_in[21];
    const float* bfc = (const float*)d_in[22];

    // ---- workspace carve-up (256B aligned) ----
    char* ws = (char*)d_ws;
    size_t off = 0;
    auto carve = [&](size_t bytes) -> char* {
        char* p = ws + off;
        off += (bytes + 255) & ~(size_t)255;
        return p;
    };
    __bf16* wswz[4];
    for (int l = 0; l < 4; ++l) wswz[l] = (__bf16*)carve((size_t)2048 * 4096 * 2);
    __bf16* wfcswz = (__bf16*)carve((size_t)1024 * (size_t)V_ * 2);
    __bf16* xh[4];
    for (int l = 0; l < 4; ++l) xh[l] = (__bf16*)carve((size_t)B_ * 2048 * 2);
    __bf16* hfc = (__bf16*)carve((size_t)B_ * 1024 * 2);
    float* cst[4];
    for (int l = 0; l < 4; ++l) cst[l] = (float*)carve((size_t)B_ * 1024 * 4);
    float* zbuf   = (float*)carve((size_t)B_ * 4096 * 4);
    float* logits = (float*)carve((size_t)B_ * (size_t)V_ * 4);
    int*   tok    = (int*)carve(256);

    float*     probs = (float*)d_out;                               // [B,T,V] f32
    long long* toks  = (long long*)(probs + (size_t)B_ * T_ * V_);  // [B,T] i64

    // ---- one-time bf16 weight swizzle (f32 -> WMMA B-fragment layout) ----
    for (int l = 0; l < 4; ++l)
        convert_weights_swz<<<(2048 * 4096) / 256, 256, 0, stream>>>(
            W[l], U[l], 1024, 2048, 4096, wswz[l]);
    convert_weights_swz<<<(1024 * V_) / 256, 256, 0, stream>>>(
        Wfc, Wfc, 1024, 1024, V_, wfcswz);

    // ---- state init ----
    for (int l = 0; l < 4; ++l)
        init_layer_state<<<(B_ * 1024) / 256, 256, 0, stream>>>(
            h_in[l], c_in[l], xh[l], cst[l]);
    init_tok<<<1, 64, 0, stream>>>(tok);

    // ---- autoregressive decode loop (stream-ordered) ----
    for (int t = 0; t < T_; ++t) {
        embed_gather<<<(B_ * 1024) / 256, 256, 0, stream>>>(emb, tok, xh[0]);
        for (int l = 0; l < 4; ++l) {
            // z = [x|h] @ [W;U]  (M=64, N=4096, K=2048): 32 WGs x 8 waves
            gemm_wmma_bf16<<<32, 256, 0, stream>>>(xh[l], wswz[l], zbuf, 4096, 2048);
            lstm_gate<<<(B_ * 1024) / 256, 256, 0, stream>>>(
                zbuf, bias[l], cst[l], xh[l], (l < 3) ? xh[l + 1] : hfc);
        }
        // logits = h4 @ Wfc  (M=64, N=32000, K=1024): 250 WGs x 8 waves
        gemm_wmma_bf16<<<250, 256, 0, stream>>>(hfc, wfcswz, logits, V_, 1024);
        softmax_argmax<<<B_, 256, 0, stream>>>(logits, bfc, probs, toks, tok, t);
    }
}
